// ScaledDotProductAttention_33904471834905
// MI455X (gfx1250) — matrix operand order repaired and checked by reading, not compiled, not run on hardware
//
#include <hip/hip_runtime.h>
#include <hip/hip_bf16.h>

typedef __attribute__((ext_vector_type(16))) _Float16 v16h;
typedef __attribute__((ext_vector_type(4)))  _Float16 v4h;
typedef __attribute__((ext_vector_type(8)))  float    v8f;

namespace cfg {
constexpr int B = 8, H = 16, S = 1024, D = 64;
constexpr float SCALE = 0.125f;          // 1/sqrt(64)
constexpr float INV_KEEP = 1.0f / 0.9f;  // dropout p = 0.1
constexpr unsigned KEEP_THRESH = 3865470566u;  // 0.9 * 2^32
}

__device__ __forceinline__ unsigned hash32(unsigned x) {
  x ^= x >> 16; x *= 0x7feb352dU;
  x ^= x >> 15; x *= 0x846ca68bU;
  x ^= x >> 16;
  return x;
}

__global__ __launch_bounds__(256)
void ScaledDotProductAttention_33904471834905_kernel(
    const float* __restrict__ Q, const float* __restrict__ Kp,
    const float* __restrict__ V, const int* __restrict__ Msk,
    float* __restrict__ Out, float* __restrict__ Attn) {
  using namespace cfg;

  __shared__ float    sc[16][S + 4];   // raw scores / exp values (fp32)
  __shared__ _Float16 pf[16][S];       // post-dropout probabilities (f16)
  __shared__ _Float16 qf[16][D];       // Q tile (f16)
  __shared__ float    part[4][16][16]; // PV partial sums (key-half 1)

  const int b  = blockIdx.z;
  const int h  = blockIdx.y;
  const int q0 = blockIdx.x * 16;
  const int tid  = threadIdx.x;
  const int wave = tid >> 5;
  const int lane = tid & 31;
  const int lm   = lane & 15;
  const bool hi  = lane >= 16;

  const size_t bh = (size_t)(b * H + h);
  const float* Qb = Q  + (bh * S + q0) * D;
  const float* Kb = Kp + bh * S * D;
  const float* Vb = V  + bh * S * D;

  // ---- Load Q tile, convert to f16 in LDS ----
  for (int idx = tid; idx < 16 * D; idx += 256) {
    int r = idx / D, d = idx % D;
    qf[r][d] = (_Float16)Qb[r * D + d];
  }
  __syncthreads();

  // ---- Build A fragments (Q rows), contiguous k layout ----
  // lane<16 : K=e
  // lane>=16: K=e+16
  v16h aq0, aq1;
#pragma unroll
  for (int e = 0; e < 16; ++e) {
    int d = (hi ? 16 : 0) + e;
    aq0[e] = qf[lm][d];
    aq1[e] = qf[lm][32 + d];
  }

  // ---- Phase 1: scores = Q * K^T, each wave does 8 key-tiles of 16 ----
  for (int kt = wave * 8; kt < wave * 8 + 8; ++kt) {
    const int k0 = kt * 16;
    // B fragment: B[d][n] = K[k0+n][d]; lane n=lm, halves cover contiguous d
    const float* krow = Kb + (size_t)(k0 + lm) * D + (hi ? 16 : 0);
    v16h b0, b1;
#pragma unroll
    for (int e = 0; e < 16; ++e) {
      b0[e] = (_Float16)krow[e];
      b1[e] = (_Float16)krow[32 + e];
    }
    v8f c = {};
    c = __builtin_amdgcn_wmma_f32_16x16x32_f16(false, aq0, false, b0,
                                               (short)0, c, false, false);
    c = __builtin_amdgcn_wmma_f32_16x16x32_f16(false, aq1, false, b1,
                                               (short)0, c, false, false);
#pragma unroll
    for (int j = 0; j < 8; ++j)
      sc[j + (hi ? 8 : 0)][k0 + lm] = c[j];  // C: M=j+8*hi, N=lm
  }

  // ---- Prefetch V panel for phase 3 (global_prefetch_b8) ----
  {
    const int r = tid;  // 256 threads x 4 rows = 1024 V rows
#pragma unroll
    for (int t = 0; t < 4; ++t) {
      const float* vrow = Vb + (size_t)(r + t * 256) * D;
      __builtin_prefetch(vrow, 0, 1);
      __builtin_prefetch(vrow + 32, 0, 1);  // second 128B of the 256B row
    }
  }
  __syncthreads();

  // ---- Phase 2: mask + softmax + dropout (4-wide vectorized) ----
  {
    const int row = tid >> 4;   // 16 threads per query row
    const int i   = tid & 15;   // thread owns k = i*4 + j*64, 4-wide chunks
    const size_t mbase = ((size_t)b * S + (q0 + row)) * S;
    const size_t abase = (bh * S + (q0 + row)) * S;
    const float NEG_INF = -__builtin_inff();

    // pass A: scale + mask, write back, row max
    float mx = NEG_INF;
#pragma unroll 4
    for (int j = 0; j < 16; ++j) {
      const int k = i * 4 + j * 64;
      float4 s = *(const float4*)&sc[row][k];
      int4   m = *(const int4*)&Msk[mbase + k];
      s.x = (m.x == 0) ? NEG_INF : s.x * SCALE;
      s.y = (m.y == 0) ? NEG_INF : s.y * SCALE;
      s.z = (m.z == 0) ? NEG_INF : s.z * SCALE;
      s.w = (m.w == 0) ? NEG_INF : s.w * SCALE;
      *(float4*)&sc[row][k] = s;
      mx = fmaxf(mx, fmaxf(fmaxf(s.x, s.y), fmaxf(s.z, s.w)));
    }
#pragma unroll
    for (int off = 8; off >= 1; off >>= 1)
      mx = fmaxf(mx, __shfl_xor(mx, off, 32));

    // pass B: exp, write back, row sum
    float sum = 0.0f;
#pragma unroll 4
    for (int j = 0; j < 16; ++j) {
      const int k = i * 4 + j * 64;
      float4 s = *(const float4*)&sc[row][k];
      s.x = __expf(s.x - mx);
      s.y = __expf(s.y - mx);
      s.z = __expf(s.z - mx);
      s.w = __expf(s.w - mx);
      *(float4*)&sc[row][k] = s;
      sum += (s.x + s.y) + (s.z + s.w);
    }
#pragma unroll
    for (int off = 8; off >= 1; off >>= 1)
      sum += __shfl_xor(sum, off, 32);

    // pass C: normalize + dropout; write attn (float4) + pf (packed f16x4)
    const float inv = 1.0f / sum;
#pragma unroll 4
    for (int j = 0; j < 16; ++j) {
      const int k = i * 4 + j * 64;
      float4 p = *(const float4*)&sc[row][k];
      p.x *= inv; p.y *= inv; p.z *= inv; p.w *= inv;
      const unsigned base = (unsigned)(abase + k);
      p.x = (hash32(base + 0) < KEEP_THRESH) ? p.x * INV_KEEP : 0.0f;
      p.y = (hash32(base + 1) < KEEP_THRESH) ? p.y * INV_KEEP : 0.0f;
      p.z = (hash32(base + 2) < KEEP_THRESH) ? p.z * INV_KEEP : 0.0f;
      p.w = (hash32(base + 3) < KEEP_THRESH) ? p.w * INV_KEEP : 0.0f;
      *(float4*)&Attn[abase + k] = p;            // post-dropout weights
      v4h hp;
      hp[0] = (_Float16)p.x; hp[1] = (_Float16)p.y;
      hp[2] = (_Float16)p.z; hp[3] = (_Float16)p.w;
      *(v4h*)&pf[row][k] = hp;                   // f16 copy for PV GEMM
    }
  }
  __syncthreads();

  // ---- Phase 3: output = P * V ; 8 waves = 4 d-tiles x 2 key-halves ----
  {
    const int ntile  = wave & 3;
    const int khalf  = wave >> 2;
    const int d0     = ntile * 16;
    const int kstart = khalf * 512;

    v8f c = {};
    for (int step = 0; step < 16; ++step) {
      const int kbase = kstart + step * 32;
      v16h a, bb;
#pragma unroll
      for (int e = 0; e < 16; ++e) {
        int kk = (hi ? 16 : 0) + e;
        a[e] = pf[lm][kbase + kk];                         // A = P rows
        int kv = (hi ? 16 : 0) + e;                        // B[k][n]=V[k][d0+n]
        bb[e] = (_Float16)Vb[(size_t)(kbase + kv) * D + d0 + lm];
      }
      c = __builtin_amdgcn_wmma_f32_16x16x32_f16(false, a, false, bb,
                                                 (short)0, c, false, false);
    }

    if (khalf == 1) {
#pragma unroll
      for (int j = 0; j < 8; ++j)
        part[ntile][j + (hi ? 8 : 0)][lm] = c[j];
    }
    __syncthreads();
    if (khalf == 0) {
      float* orow = Out + (bh * S + q0) * D;
#pragma unroll
      for (int j = 0; j < 8; ++j) {
        int m = j + (hi ? 8 : 0);
        orow[(size_t)m * D + d0 + lm] = c[j] + part[ntile][m][lm];
      }
    }
  }
}

extern "C" void kernel_launch(void* const* d_in, const int* in_sizes, int n_in,
                              void* d_out, int out_size, void* d_ws, size_t ws_size,
                              hipStream_t stream) {
  using namespace cfg;
  (void)in_sizes; (void)n_in; (void)out_size; (void)d_ws; (void)ws_size;

  const float* Q   = (const float*)d_in[0];
  const float* K   = (const float*)d_in[1];
  const float* V   = (const float*)d_in[2];
  const int*   Msk = (const int*)d_in[3];

  float* out  = (float*)d_out;                       // [B,H,S,D] first
  float* attn = out + (size_t)B * H * S * D;         // then [B,H,S,S]

  dim3 grid(S / 16, H, B);   // 64 x 16 x 8 = 8192 workgroups
  dim3 block(256);           // 8 waves (wave32)
  ScaledDotProductAttention_33904471834905_kernel<<<grid, block, 0, stream>>>(
      Q, K, V, Msk, out, attn);
}
